// KNRM_61993557950578
// MI455X (gfx1250) — matrix operand order repaired and checked by reading, not compiled, not run on hardware
//
#include <hip/hip_runtime.h>
#include <hip/hip_bf16.h>
#include <math.h>

// ---------------------------------------------------------------------------
// KNRM fused kernel for MI455X (gfx1250, wave32, WMMA).
//
// Roofline: gathers ~300MB (~13us @ 23.3TB/s, emb table resident in 192MB L2).
// GEMM is only 8.6 GFLOP -> v_wmma_f32_16x16x32_f16, negligible even when
// recomputed 3x. True limiter: ~1.4G v_exp_f32 for the 21 RBF kernels.
// Therefore: (a) keep mm tiles in registers (never spill the 33M-element
// similarity matrix), (b) minimize VALU per exp via exp2(-(a*x+b)^2) with a
// free neg-source-modifier on v_exp_f32, (c) keep VGPRs < 256 (3 kernel
// groups of 7) so no s_set_vgpr_msb/SALU overhead and high wave occupancy to
// hide the TRANS pipeline.
// ---------------------------------------------------------------------------

typedef __attribute__((ext_vector_type(16))) _Float16 v16h;
typedef __attribute__((ext_vector_type(8)))  float    v8f;

#define BATCH 1024
#define LQ    64
#define LD    512
#define DIM   64
#define NK    21
#define NG    3    // kernel groups
#define GK    7    // kernels per group
#define QSTR  68   // padded f16 row stride (34 dwords -> conflict-free)
#define DSTR  68

// exp(-0.5*((x-mu)/sg)^2) == exp2(-(a*x+b)^2),  a = (1/sg)*sqrt(0.5*log2(e)),
// b = -mu*a.  sqrt(0.5*log2(e)) = 0.84932439e0.
#define SC 0.8493243922672166f
__constant__ float c_a[NK] = {
    10*SC, 10*SC, 10*SC, 10*SC, 10*SC, 10*SC, 10*SC, 10*SC, 10*SC, 10*SC,
    10*SC, 10*SC, 10*SC, 10*SC, 10*SC, 10*SC, 10*SC, 10*SC, 10*SC, 10*SC,
    1000*SC };
__constant__ float c_b[NK] = {
     0.95f*10*SC,  0.85f*10*SC,  0.75f*10*SC,  0.65f*10*SC,  0.55f*10*SC,
     0.45f*10*SC,  0.35f*10*SC,  0.25f*10*SC,  0.15f*10*SC,  0.05f*10*SC,
    -0.05f*10*SC, -0.15f*10*SC, -0.25f*10*SC, -0.35f*10*SC, -0.45f*10*SC,
    -0.55f*10*SC, -0.65f*10*SC, -0.75f*10*SC, -0.85f*10*SC, -0.95f*10*SC,
    -1.00f*1000*SC };

// B fragment (32x16 f16 per K-block): lane%16 = column (doc row),
// VGPR v holds K = (v<4?0:16) + (lane/16)*8 + (v&3)*2, +1 (same order as A).
__device__ __forceinline__ void load_bfrag(const _Float16* __restrict__ dsh,
                                           int nt, int lrow, int half,
                                           v16h (&bf)[2])
{
    #pragma unroll
    for (int kb = 0; kb < 2; ++kb) {
        #pragma unroll
        for (int v = 0; v < 8; ++v) {
            int k = kb * 32 + ((v & 4) << 2) + half * 8 + (v & 3) * 2;
            const _Float16* src = &dsh[(nt * 16 + lrow) * DSTR + k];
            bf[kb][2 * v + 0] = src[0];
            bf[kb][2 * v + 1] = src[1];
        }
    }
}

__global__ __launch_bounds__(256)
void knrm_pair_kernel(const int* __restrict__ q1, const int* __restrict__ d1,
                      const int* __restrict__ q2, const int* __restrict__ d2,
                      const float* __restrict__ emb,
                      const float* __restrict__ w1, const float* __restrict__ b1,
                      const float* __restrict__ w2, const float* __restrict__ b2,
                      const float* __restrict__ w3, const float* __restrict__ b3,
                      float* __restrict__ logits)
{
    extern __shared__ char smem[];
    _Float16* qsh    = (_Float16*)smem;            // [LQ][QSTR] normalized query (f16)
    _Float16* dsh    = qsh + LQ * QSTR;            // [LD][DSTR] normalized doc   (f16)
    float*    rowsum = (float*)(dsh + LD * DSTR);  // [LQ][NK]
    float*    feats  = rowsum + LQ * NK;           // [NK]

    const int tid  = threadIdx.x;
    const int lane = tid & 31;
    const int w    = tid >> 5;          // 8 waves / block
    const int p    = blockIdx.x >> 10;  // pair select
    const int b    = blockIdx.x & (BATCH - 1);

    const int* qtok = (p == 0 ? q1 : q2) + b * LQ;
    const int* dtok = (p == 0 ? d1 : d2) + b * LD;

    for (int i = tid; i < LQ * NK; i += 256) rowsum[i] = 0.0f;

    // ---- Phase 1: gather + L2-normalize -> f16 LDS (one wave per row) ----
    for (int r = w; r < LQ; r += 8) {
        const float* e = emb + (size_t)qtok[r] * DIM;
        float x0 = e[lane * 2 + 0], x1 = e[lane * 2 + 1];
        float ss = x0 * x0 + x1 * x1;
        #pragma unroll
        for (int off = 16; off > 0; off >>= 1) ss += __shfl_xor(ss, off, 32);
        float inv = 1.0f / fmaxf(sqrtf(ss), 1e-12f);
        qsh[r * QSTR + lane * 2 + 0] = (_Float16)(x0 * inv);
        qsh[r * QSTR + lane * 2 + 1] = (_Float16)(x1 * inv);
    }
    for (int r = w; r < LD; r += 8) {
        const float* e = emb + (size_t)dtok[r] * DIM;
        float x0 = e[lane * 2 + 0], x1 = e[lane * 2 + 1];
        float ss = x0 * x0 + x1 * x1;
        #pragma unroll
        for (int off = 16; off > 0; off >>= 1) ss += __shfl_xor(ss, off, 32);
        float inv = 1.0f / fmaxf(sqrtf(ss), 1e-12f);
        dsh[r * DSTR + lane * 2 + 0] = (_Float16)(x0 * inv);
        dsh[r * DSTR + lane * 2 + 1] = (_Float16)(x1 * inv);
    }
    __syncthreads();

    // ---- Phase 2: WMMA GEMM tiles + fused RBF accumulation -------------
    const int mt   = w >> 1;       // M-tile 0..3 (2 waves per M-tile)
    const int half = lane >> 4;    // lane half (ISA A/B/C layouts)
    const int lrow = lane & 15;
    const int nt0  = (w & 1);

    // A fragment (16x32 f16 per kb): VGPR v <- K = (v<4?0:16) + half*8 + (v&3)*2
    v16h afrag[2];
    #pragma unroll
    for (int kb = 0; kb < 2; ++kb) {
        #pragma unroll
        for (int v = 0; v < 8; ++v) {
            int k = kb * 32 + ((v & 4) << 2) + half * 8 + (v & 3) * 2;
            const _Float16* src = &qsh[(mt * 16 + lrow) * QSTR + k];
            afrag[kb][2 * v + 0] = src[0];
            afrag[kb][2 * v + 1] = src[1];
        }
    }

    // 3 passes of 7 RBF kernels: acc stays at 56 VGPRs -> whole wave < 256.
    for (int g = 0; g < NG; ++g) {
        float acc[8][GK];
        #pragma unroll
        for (int v = 0; v < 8; ++v)
            #pragma unroll
            for (int jj = 0; jj < GK; ++jj) acc[v][jj] = 0.0f;

        float ga[GK], gb[GK];
        #pragma unroll
        for (int jj = 0; jj < GK; ++jj) {
            ga[jj] = c_a[g * GK + jj];
            gb[jj] = c_b[g * GK + jj];
        }

        v16h bf[2];
        load_bfrag(dsh, nt0, lrow, half, bf);

        #pragma unroll 2
        for (int nt = nt0; nt < 32; nt += 2) {
            v8f c = {};
            c = __builtin_amdgcn_wmma_f32_16x16x32_f16(false, afrag[0], false, bf[0],
                                                       (short)0, c, false, false);
            c = __builtin_amdgcn_wmma_f32_16x16x32_f16(false, afrag[1], false, bf[1],
                                                       (short)0, c, false, false);
            // prefetch next tile's B fragment; exp stage below hides DS latency
            int ntn = (nt + 2 < 32) ? nt + 2 : nt0;
            v16h bn[2];
            load_bfrag(dsh, ntn, lrow, half, bn);

            // C layout: VGPR v = row (mt*16 + v + 8*half), lane%16 = doc column.
            #pragma unroll
            for (int v = 0; v < 8; ++v) {
                float x = c[v];
                #pragma unroll
                for (int jj = 0; jj < GK; ++jj) {
                    float t = fmaf(x, ga[jj], gb[jj]);
                    acc[v][jj] += __builtin_amdgcn_exp2f(-(t * t)); // v_exp_f32, neg mod
                }
            }
            bf[0] = bn[0];
            bf[1] = bn[1];
        }

        // reduce over the 16 doc-lanes in each half, then combine the 2 waves
        // sharing this M-tile via LDS float atomics (ds_add_f32).
        #pragma unroll
        for (int v = 0; v < 8; ++v) {
            #pragma unroll
            for (int jj = 0; jj < GK; ++jj) {
                float s = acc[v][jj];
                s += __shfl_xor(s, 1, 32);
                s += __shfl_xor(s, 2, 32);
                s += __shfl_xor(s, 4, 32);
                s += __shfl_xor(s, 8, 32);
                if (lrow == 0)
                    atomicAdd(&rowsum[(mt * 16 + half * 8 + v) * NK + g * GK + jj], s);
            }
        }
    }
    __syncthreads();

    // ---- Phase 3: log1p over doc-sums, sum over query rows ----
    if (tid < NK) {
        float f = 0.0f;
        for (int r = 0; r < LQ; ++r) f += __logf(1.0f + rowsum[r * NK + tid]);
        feats[tid] = f;
    }
    __syncthreads();

    // ---- Phase 4: tiny MLP 21 -> 10 -> 5 -> 1 ----
    if (tid == 0) {
        float h1[10], h2[5];
        #pragma unroll
        for (int i = 0; i < 10; ++i) {
            float s = b1[i];
            #pragma unroll
            for (int j = 0; j < NK; ++j) s += w1[i * NK + j] * feats[j];
            h1[i] = fmaxf(s, 0.0f);
        }
        #pragma unroll
        for (int i = 0; i < 5; ++i) {
            float s = b2[i];
            #pragma unroll
            for (int j = 0; j < 10; ++j) s += w2[i * 10 + j] * h1[j];
            h2[i] = fmaxf(s, 0.0f);
        }
        float l = b3[0];
        #pragma unroll
        for (int j = 0; j < 5; ++j) l += w3[j] * h2[j];
        logits[p * BATCH + b] = l;
    }
}

__global__ void knrm_sigmoid_kernel(const float* __restrict__ logits,
                                    float* __restrict__ out)
{
    int b = blockIdx.x * blockDim.x + threadIdx.x;
    if (b < BATCH) {
        float d = logits[b] - logits[BATCH + b];
        out[b] = 1.0f / (1.0f + __expf(-d));
    }
}

extern "C" void kernel_launch(void* const* d_in, const int* in_sizes, int n_in,
                              void* d_out, int out_size, void* d_ws, size_t ws_size,
                              hipStream_t stream)
{
    const int*   q1  = (const int*)d_in[0];
    const int*   d1  = (const int*)d_in[1];
    const int*   q2  = (const int*)d_in[2];
    const int*   d2  = (const int*)d_in[3];
    const float* emb = (const float*)d_in[4];
    const float* w1  = (const float*)d_in[5];
    const float* b1  = (const float*)d_in[6];
    const float* w2  = (const float*)d_in[7];
    const float* b2  = (const float*)d_in[8];
    const float* w3  = (const float*)d_in[9];
    const float* b3  = (const float*)d_in[10];

    float* logits = (float*)d_ws;   // 2 * 1024 floats

    size_t smem = (size_t)(LQ * QSTR + LD * DSTR) * sizeof(_Float16)
                + (size_t)(LQ * NK + NK) * sizeof(float);   // ~84 KB (WGP has 320 KB)

    knrm_pair_kernel<<<dim3(2 * BATCH), dim3(256), smem, stream>>>(
        q1, d1, q2, d2, emb, w1, b1, w2, b2, w3, b3, logits);

    knrm_sigmoid_kernel<<<dim3(4), dim3(256), 0, stream>>>(logits, (float*)d_out);
}
